// decoderRNN_30133490549688
// MI455X (gfx1250) — compile-verified, matrix operand order checked by reading, not run on hardware
//
#include <hip/hip_runtime.h>
#include <hip/hip_bf16.h>

// ---------------------------------------------------------------------------
// Problem constants (from reference)
// ---------------------------------------------------------------------------
#define BB 64
#define LL 49
#define FF 2048
#define TT 32
#define VV 15000
#define EE 512
#define HH 512
#define AA 512
#define XK (EE + FF + HH)   // 3072 : [emb | context | h] -> fused LSTM input
#define G4 (4 * HH)         // 2048 : gate rows

typedef __attribute__((ext_vector_type(16))) __bf16        v16bf;
typedef __attribute__((ext_vector_type(8)))  float         v8f;
typedef __attribute__((ext_vector_type(4)))  unsigned int  u32x4;

// One WMMA operand fragment: 16 bf16 per lane = two 16B contiguous runs.
struct Frag { u32x4 lo, hi; };

__device__ __forceinline__ v16bf frag_cast(Frag f) {
    union { Frag f; v16bf v; } u;
    u.f = f;
    return u.v;
}

__device__ __forceinline__ Frag ldfrag(const unsigned short* p) {
    // p is 16B aligned: lda even, k-block multiple of 32, grp*8 elems = 16B
    Frag f;
    f.lo = *(const u32x4*)(p);
    f.hi = *(const u32x4*)(p + 16);
    return f;
}

__device__ __forceinline__ unsigned short f2bf(float f) {
    unsigned int x = __builtin_bit_cast(unsigned int, f);
    x += 0x7FFFu + ((x >> 16) & 1u);          // round-to-nearest-even
    return (unsigned short)(x >> 16);
}

__device__ __forceinline__ float sigmoidf_(float x) {
    return 1.0f / (1.0f + __expf(-x));
}

// ---------------------------------------------------------------------------
// Generic bf16 WMMA GEMM:  D[m,n] = sum_k A[m,k] * Bw[n,k] + bias[n]
//   A : [M, K] bf16 row-major (M = gridDim.y*64, multiple of 64)
//   Bw: [nrows, K] bf16 row-major (torch Linear weight convention)
//   D : fp32, row stride ldd
// Block: 128 threads (4 waves), tile 64x64, wave tile 16x64.
// No LDS: fragments loaded straight from global (L2-resident operands);
// per CDNA5 ISA 7.12.2 a lane's 16-bit A/B fragment is two contiguous 16B
// runs at k = grp*8 and k = 16+grp*8 -> two global_load_b128 per fragment.
// Out-of-range B rows (N tail) are clamped to row 0: B-lane n only feeds
// D column n, and those columns are masked at the store.
// ---------------------------------------------------------------------------
__global__ __launch_bounds__(128) void wmma_gemm_bf16(
    const unsigned short* __restrict__ A, int lda,
    const unsigned short* __restrict__ Bw, int ldb, int nrows,
    const float* __restrict__ bias,
    float* __restrict__ D, int ldd,
    int K)
{
    const int tid   = threadIdx.x;
    const int n0    = blockIdx.x * 64;
    const int m0    = blockIdx.y * 64;
    const int wave  = tid >> 5;
    const int lane  = tid & 31;
    const int lrow  = lane & 15;
    const int grp   = lane >> 4;
    const int m_off = wave * 16;

    const unsigned short* arow = A + (size_t)(m0 + m_off + lrow) * lda + grp * 8;
    const unsigned short* brow[4];
#pragma unroll
    for (int nt = 0; nt < 4; ++nt) {
        int r = n0 + nt * 16 + lrow;
        if (r >= nrows) r = 0;               // clamp: column masked at store
        brow[nt] = Bw + (size_t)r * ldb + grp * 8;
    }

    v8f zero = {};
    v8f acc[4];
#pragma unroll
    for (int nt = 0; nt < 4; ++nt) acc[nt] = zero;

    // software pipeline: load k-block (kb) while multiplying k-block (kb-32)
    Frag fa = ldfrag(arow);
    Frag fb[4];
#pragma unroll
    for (int nt = 0; nt < 4; ++nt) fb[nt] = ldfrag(brow[nt]);

    for (int kb = 32; kb < K; kb += 32) {
        Frag na = ldfrag(arow + kb);
        Frag nb[4];
#pragma unroll
        for (int nt = 0; nt < 4; ++nt) nb[nt] = ldfrag(brow[nt] + kb);

        v16bf av = frag_cast(fa);
#pragma unroll
        for (int nt = 0; nt < 4; ++nt)
            acc[nt] = __builtin_amdgcn_wmma_f32_16x16x32_bf16(
                false, av, false, frag_cast(fb[nt]), (short)0, acc[nt],
                false, false);

        fa = na;
#pragma unroll
        for (int nt = 0; nt < 4; ++nt) fb[nt] = nb[nt];
    }
    {   // epilogue k-block
        v16bf av = frag_cast(fa);
#pragma unroll
        for (int nt = 0; nt < 4; ++nt)
            acc[nt] = __builtin_amdgcn_wmma_f32_16x16x32_bf16(
                false, av, false, frag_cast(fb[nt]), (short)0, acc[nt],
                false, false);
    }

    // store D: C/D layout -> lane n = lane&15, VGPR r -> m = r + 8*(lane>>4)
#pragma unroll
    for (int nt = 0; nt < 4; ++nt) {
        int n = n0 + nt * 16 + lrow;
        if (n < nrows) {
            float bval = bias ? bias[n] : 0.0f;
#pragma unroll
            for (int r = 0; r < 8; ++r) {
                int m = m0 + m_off + r + 8 * grp;
                D[(size_t)m * ldd + n] = acc[nt][r] + bval;
            }
        }
    }
}

// ---------------------------------------------------------------------------
// one-time prep kernels
// ---------------------------------------------------------------------------
__global__ void cvt_bf16_kernel(const float* __restrict__ src,
                                unsigned short* __restrict__ dst, int n) {
    for (int i = blockIdx.x * blockDim.x + threadIdx.x; i < n;
         i += gridDim.x * blockDim.x)
        dst[i] = f2bf(src[i]);
}

// Wcat[n][k] : k<2560 -> W_ih[n][k] ; else W_hh[n][k-2560]
__global__ void build_wcat_kernel(const float* __restrict__ W_ih,
                                  const float* __restrict__ W_hh,
                                  unsigned short* __restrict__ Wcat) {
    const int total = G4 * XK;
    for (int i = blockIdx.x * blockDim.x + threadIdx.x; i < total;
         i += gridDim.x * blockDim.x) {
        int n = i / XK, k = i - n * XK;
        float v = (k < EE + FF) ? W_ih[(size_t)n * (EE + FF) + k]
                                : W_hh[(size_t)n * HH + (k - (EE + FF))];
        Wcat[i] = f2bf(v);
    }
}

__global__ void bias_sum_kernel(const float* __restrict__ a,
                                const float* __restrict__ b,
                                float* __restrict__ out, int n) {
    int i = blockIdx.x * blockDim.x + threadIdx.x;
    if (i < n) out[i] = a[i] + b[i];
}

// mean over L of features -> bf16 [B, F]
__global__ void feat_mean_kernel(const float* __restrict__ feat,
                                 unsigned short* __restrict__ mean_bf) {
    const int total = BB * FF;
    for (int i = blockIdx.x * blockDim.x + threadIdx.x; i < total;
         i += gridDim.x * blockDim.x) {
        int b = i >> 11, f = i & (FF - 1);
        const float* p = feat + ((size_t)b * LL) * FF + f;
        float s = 0.0f;
        for (int l = 0; l < LL; ++l) s += p[(size_t)l * FF];
        mean_bf[i] = f2bf(s * (1.0f / (float)LL));
    }
}

// ---------------------------------------------------------------------------
// per-step attention: att2 = h@U_a^T + b ; score = tanh(att1+att2)@v_a + b ;
// softmax ; context = w @ features ; assemble x_cat = [emb | context | h] bf16
// one block per batch element, 256 threads
// ---------------------------------------------------------------------------
__global__ __launch_bounds__(256) void attn_step_kernel(
    const float* __restrict__ features,     // [B,L,F]
    const float* __restrict__ att1,         // [B,L,A]
    const float* __restrict__ U_a_W,        // [A,H]
    const float* __restrict__ U_a_b,        // [A]
    const float* __restrict__ v_a_W,        // [1,A]
    const float* __restrict__ v_a_b,        // [1]
    const float* __restrict__ h,            // [B,H]
    const float* __restrict__ embed,        // [V,E]
    const int*   __restrict__ captions,     // [B,T]
    unsigned short* __restrict__ x_cat,     // [B, XK] bf16
    int t)
{
    __shared__ float sh_h[HH];
    __shared__ float sh_att2[AA];
    __shared__ float sh_score[64];
    __shared__ float red[8];

    const int b   = blockIdx.x;
    const int tid = threadIdx.x;

    for (int j = tid; j < HH; j += 256) sh_h[j] = h[b * HH + j];
    __syncthreads();

    // att2 (fp32 VALU: tiny, 0.5 MFLOP/block)
    for (int j = tid; j < AA; j += 256) {
        const float* w = U_a_W + (size_t)j * HH;
        float acc = U_a_b[j];
        for (int k = 0; k < HH; ++k) acc = fmaf(sh_h[k], w[k], acc);
        sh_att2[j] = acc;
    }
    __syncthreads();

    // scores over L
    const float* a1b = att1 + (size_t)b * LL * AA;
    for (int l = 0; l < LL; ++l) {
        float p = 0.0f;
        for (int a = tid; a < AA; a += 256)
            p += tanhf(a1b[(size_t)l * AA + a] + sh_att2[a]) * v_a_W[a];
        for (int off = 16; off > 0; off >>= 1) p += __shfl_down(p, off, 32);
        if ((tid & 31) == 0) red[tid >> 5] = p;
        __syncthreads();
        if (tid == 0) {
            float s = v_a_b[0];
            for (int w = 0; w < 8; ++w) s += red[w];
            sh_score[l] = s;
        }
        __syncthreads();
    }

    // softmax over 49 (serial: negligible)
    if (tid == 0) {
        float mx = sh_score[0];
        for (int l = 1; l < LL; ++l) mx = fmaxf(mx, sh_score[l]);
        float sum = 0.0f;
        for (int l = 0; l < LL; ++l) { float e = __expf(sh_score[l] - mx); sh_score[l] = e; sum += e; }
        float inv = 1.0f / sum;
        for (int l = 0; l < LL; ++l) sh_score[l] *= inv;
    }
    __syncthreads();

    // context -> x_cat[E .. E+F)
    unsigned short* xb = x_cat + (size_t)b * XK;
    const float* fb = features + (size_t)b * LL * FF;
    for (int f = tid; f < FF; f += 256) {
        float acc = 0.0f;
        for (int l = 0; l < LL; ++l) acc = fmaf(sh_score[l], fb[(size_t)l * FF + f], acc);
        xb[EE + f] = f2bf(acc);
    }
    // embedding -> x_cat[0..E)
    int tok = captions[b * TT + t];
    for (int e = tid; e < EE; e += 256)
        xb[e] = f2bf(embed[(size_t)tok * EE + e]);
    // h -> x_cat[E+F .. XK)
    for (int j = tid; j < HH; j += 256)
        xb[EE + FF + j] = f2bf(sh_h[j]);
}

// ---------------------------------------------------------------------------
// LSTM pointwise: gates [B, 4H] (i|f|g|o) -> h,c update + h_bf
// ---------------------------------------------------------------------------
__global__ void lstm_pointwise_kernel(const float* __restrict__ gates,
                                      float* __restrict__ h,
                                      float* __restrict__ c,
                                      unsigned short* __restrict__ h_bf) {
    int idx = blockIdx.x * blockDim.x + threadIdx.x;
    if (idx >= BB * HH) return;
    int b = idx >> 9, j = idx & (HH - 1);
    const float* g = gates + (size_t)b * G4;
    float gi = sigmoidf_(g[j]);
    float gf = sigmoidf_(g[HH + j]);
    float gg = tanhf(g[2 * HH + j]);
    float go = sigmoidf_(g[3 * HH + j]);
    float cn = gf * c[idx] + gi * gg;
    float hn = go * tanhf(cn);
    c[idx] = cn;
    h[idx] = hn;
    h_bf[idx] = f2bf(hn);
}

// ---------------------------------------------------------------------------
// host orchestration
// ---------------------------------------------------------------------------
extern "C" void kernel_launch(void* const* d_in, const int* in_sizes, int n_in,
                              void* d_out, int out_size, void* d_ws, size_t ws_size,
                              hipStream_t stream) {
    (void)in_sizes; (void)n_in; (void)out_size; (void)ws_size;

    const float* features = (const float*)d_in[0];
    const int*   captions = (const int*)d_in[1];
    // d_in[2] = force_prob (== 1, teacher forcing) -> unused
    const float* init_h_W = (const float*)d_in[3];
    const float* init_h_b = (const float*)d_in[4];
    const float* init_c_W = (const float*)d_in[5];
    const float* init_c_b = (const float*)d_in[6];
    const float* W_a_W    = (const float*)d_in[7];
    const float* W_a_b    = (const float*)d_in[8];
    const float* U_a_W    = (const float*)d_in[9];
    const float* U_a_b    = (const float*)d_in[10];
    const float* v_a_W    = (const float*)d_in[11];
    const float* v_a_b    = (const float*)d_in[12];
    const float* embed    = (const float*)d_in[13];
    const float* W_ih     = (const float*)d_in[14];
    const float* W_hh     = (const float*)d_in[15];
    const float* b_ih     = (const float*)d_in[16];
    const float* b_hh     = (const float*)d_in[17];
    const float* fc_W     = (const float*)d_in[18];
    const float* fc_b     = (const float*)d_in[19];
    float* out = (float*)d_out;

    // workspace carve-up (256B aligned)
    char* w = (char*)d_ws;
    size_t off = 0;
    auto alloc = [&](size_t bytes) -> void* {
        void* p = w + off;
        off += (bytes + 255) & ~(size_t)255;
        return p;
    };
    unsigned short* features_bf = (unsigned short*)alloc((size_t)BB * LL * FF * 2);
    unsigned short* W_a_bf      = (unsigned short*)alloc((size_t)AA * FF * 2);
    unsigned short* init_h_bf   = (unsigned short*)alloc((size_t)HH * FF * 2);
    unsigned short* init_c_bf   = (unsigned short*)alloc((size_t)HH * FF * 2);
    unsigned short* Wcat_bf     = (unsigned short*)alloc((size_t)G4 * XK * 2);
    unsigned short* fc_bf       = (unsigned short*)alloc((size_t)VV * HH * 2);
    unsigned short* mean_bf     = (unsigned short*)alloc((size_t)BB * FF * 2);
    unsigned short* x_cat       = (unsigned short*)alloc((size_t)BB * XK * 2);
    unsigned short* h_bf        = (unsigned short*)alloc((size_t)BB * HH * 2);
    float* att1      = (float*)alloc((size_t)BB * LL * AA * 4);
    float* gates     = (float*)alloc((size_t)BB * G4 * 4);
    float* h_buf     = (float*)alloc((size_t)BB * HH * 4);
    float* c_buf     = (float*)alloc((size_t)BB * HH * 4);
    float* gate_bias = (float*)alloc((size_t)G4 * 4);

    // ---- one-time precompute ----
    cvt_bf16_kernel<<<1024, 256, 0, stream>>>(features, features_bf, BB * LL * FF);
    cvt_bf16_kernel<<<512, 256, 0, stream>>>(W_a_W, W_a_bf, AA * FF);
    cvt_bf16_kernel<<<512, 256, 0, stream>>>(init_h_W, init_h_bf, HH * FF);
    cvt_bf16_kernel<<<512, 256, 0, stream>>>(init_c_W, init_c_bf, HH * FF);
    cvt_bf16_kernel<<<1024, 256, 0, stream>>>(fc_W, fc_bf, VV * HH);
    build_wcat_kernel<<<1024, 256, 0, stream>>>(W_ih, W_hh, Wcat_bf);
    bias_sum_kernel<<<(G4 + 255) / 256, 256, 0, stream>>>(b_ih, b_hh, gate_bias, G4);
    feat_mean_kernel<<<512, 256, 0, stream>>>(features, mean_bf);

    // h0 = mean_f @ init_h_W^T + b : [64,2048]x[2048->512]
    wmma_gemm_bf16<<<dim3(HH / 64, BB / 64), 128, 0, stream>>>(
        mean_bf, FF, init_h_bf, FF, HH, init_h_b, h_buf, HH, FF);
    wmma_gemm_bf16<<<dim3(HH / 64, BB / 64), 128, 0, stream>>>(
        mean_bf, FF, init_c_bf, FF, HH, init_c_b, c_buf, HH, FF);

    // att1 = features @ W_a^T + b : [3136,2048]x[2048->512]
    wmma_gemm_bf16<<<dim3(AA / 64, (BB * LL) / 64), 128, 0, stream>>>(
        features_bf, FF, W_a_bf, FF, AA, W_a_b, att1, AA, FF);

    // ---- time loop (teacher forcing) ----
    for (int t = 0; t < TT; ++t) {
        attn_step_kernel<<<BB, 256, 0, stream>>>(
            features, att1, U_a_W, U_a_b, v_a_W, v_a_b,
            h_buf, embed, captions, x_cat, t);

        // gates = x_cat @ Wcat^T + (b_ih+b_hh) : [64,3072]x[3072->2048]
        wmma_gemm_bf16<<<dim3(G4 / 64, BB / 64), 128, 0, stream>>>(
            x_cat, XK, Wcat_bf, XK, G4, gate_bias, gates, G4, XK);

        lstm_pointwise_kernel<<<(BB * HH + 255) / 256, 256, 0, stream>>>(
            gates, h_buf, c_buf, h_bf);

        // logits_t = h_new @ fc_W^T + fc_b : [64,512]x[512->15000]
        // write straight into out[b, t, v]  (row stride T*V, col offset t*V)
        wmma_gemm_bf16<<<dim3((VV + 63) / 64, BB / 64), 128, 0, stream>>>(
            h_bf, HH, fc_bf, HH, VV, fc_b, out + (size_t)t * VV, TT * VV, HH);
    }
}
